// WordAspectFusion_91001767068125
// MI455X (gfx1250) — matrix operand order repaired and checked by reading, not compiled, not run on hardware
//
#include <hip/hip_runtime.h>

typedef __attribute__((ext_vector_type(16))) __bf16 v16bf;
typedef __attribute__((ext_vector_type(8)))  float  v8f;
typedef int v4i __attribute__((vector_size(16)));   // matches builtin's V4i pointee

#define NB   32      // batches
#define SD   512     // sequence (M)
#define DD   1024    // feature dim (K == N)
#define BM   128     // block tile M
#define BN   128     // block tile N
#define KT   32      // K step (one bf16 WMMA depth)
#define LDSP 48      // padded LDS row stride in elements (96B, keeps b128 alignment)

#if defined(__has_builtin)
# if __has_builtin(__builtin_amdgcn_global_load_async_to_lds_b128) && \
     __has_builtin(__builtin_amdgcn_s_wait_asynccnt)
#  define HAVE_ASYNC 1
# endif
#endif
#ifndef HAVE_ASYNC
# define HAVE_ASYNC 0
#endif

#define AS1 __attribute__((address_space(1)))
#define AS3 __attribute__((address_space(3)))

union FragB { uint4 q[2]; v16bf v; };

__device__ __forceinline__ unsigned short f32_to_bf16_rn(float x) {
    unsigned int u = __float_as_uint(x);
    unsigned int r = u + 0x7FFFu + ((u >> 16) & 1u);   // round to nearest even
    return (unsigned short)(r >> 16);
}

__device__ __forceinline__ void split_bf16(float x, unsigned short &hi, unsigned short &lo) {
    hi = f32_to_bf16_rn(x);
    float hf = __uint_as_float(((unsigned int)hi) << 16);
    lo = f32_to_bf16_rn(x - hf);                        // exact residual capture
}

__global__ __launch_bounds__(256)
void wordaspect_circgemm(const float* __restrict__ emb,
                         const float* __restrict__ aspect,
                         float* __restrict__ out)
{
    __shared__ __attribute__((aligned(16))) unsigned short sAh[BM * LDSP];
    __shared__ __attribute__((aligned(16))) unsigned short sAl[BM * LDSP];
    __shared__ __attribute__((aligned(16))) unsigned short sBh[BN * LDSP];
    __shared__ __attribute__((aligned(16))) unsigned short sBl[BN * LDSP];
    __shared__ __attribute__((aligned(16))) unsigned short sCh[2 * DD];  // circular ahat hi
    __shared__ __attribute__((aligned(16))) unsigned short sCl[2 * DD];  // circular ahat lo
    __shared__ float sRed[256];
    __shared__ float sRn[BM];
#if HAVE_ASYNC
    __shared__ __attribute__((aligned(16))) float sRaw[2][BM * KT];      // async raw f32 staging
#endif

    const int tid = threadIdx.x;
    const int b   = blockIdx.z;
    const int bm0 = blockIdx.x * BM;
    const int bn0 = blockIdx.y * BN;

    // ---------- Phase 1: normalize aspect[b,:], build duplicated circular bf16 hi/lo ----------
    {
        const float4 a4 = *(const float4*)(aspect + (size_t)b * DD + tid * 4);
        float av0 = a4.x, av1 = a4.y, av2 = a4.z, av3 = a4.w;
        sRed[tid] = av0 * av0 + av1 * av1 + av2 * av2 + av3 * av3;
        __syncthreads();
        #pragma unroll
        for (int off = 128; off > 0; off >>= 1) {
            if (tid < off) sRed[tid] += sRed[tid + off];
            __syncthreads();
        }
        float sum = sRed[0];
        float rn = (sum > 0.f && sum <= 3.38e38f) ? rsqrtf(sum) : 0.f;
        __syncthreads();   // release sRed for phase 2
        float vv[4] = {av0 * rn, av1 * rn, av2 * rn, av3 * rn};
        #pragma unroll
        for (int j = 0; j < 4; ++j) {
            float v = vv[j];
            v = (v == v) ? v : 0.f;            // NaN -> 0 (matches reference)
            unsigned short h, l;
            split_bf16(v, h, l);
            int d = tid * 4 + j;
            sCh[d] = h; sCh[d + DD] = h;
            sCl[d] = l; sCl[d + DD] = l;
        }
    }

    // ---------- Phase 2: inverse L2 norms for this block's 128 embedding rows ----------
    {
        const int r = tid >> 1, half = tid & 1;
        const float* rp = emb + ((size_t)(b * SD + bm0 + r)) * DD + half * (DD / 2);
        float ss = 0.f;
        #pragma unroll 4
        for (int j = 0; j < DD / 2; j += 4) {
            float4 v = *(const float4*)(rp + j);
            ss += v.x * v.x + v.y * v.y + v.z * v.z + v.w * v.w;
        }
        sRed[tid] = ss;
        __syncthreads();
        if (tid < BM) {
            float sum = sRed[2 * tid] + sRed[2 * tid + 1];
            sRn[tid] = (sum > 0.f && sum <= 3.38e38f) ? rsqrtf(sum) : 0.f;
        }
        __syncthreads();
    }

    // ---------- Phase 3: K-loop GEMM with bf16x2 compensated WMMA ----------
    const int lane = tid & 31;
    const int wv   = tid >> 5;       // 8 waves
    const int wm   = wv & 3;         // wave M offset: wm*32
    const int wn   = wv >> 2;        // wave N offset: wn*64
    const int lr   = lane & 15;
    const int lh   = lane >> 4;      // lane half (0/1)

#if HAVE_ASYNC
    // async DMA of one raw f32 A tile (128x32 = 16KB) into LDS staging buffer
    auto issue_tile = [&](int ktn, int buf) {
        const int r = tid >> 1, half = tid & 1;
        const float* g = emb + ((size_t)(b * SD + bm0 + r)) * DD + ktn + half * 16;
        float* l = &sRaw[buf][r * KT + half * 16];
        #pragma unroll
        for (int j = 0; j < 4; ++j) {
            __builtin_amdgcn_global_load_async_to_lds_b128(
                (AS1 v4i*)(g + 4 * j), (AS3 v4i*)(l + 4 * j), 0, 0);
        }
    };
    issue_tile(0, 0);
    __builtin_amdgcn_s_wait_asynccnt(0);
    __syncthreads();
#endif

    v8f acc[2][4];
    #pragma unroll
    for (int i = 0; i < 2; ++i)
        #pragma unroll
        for (int j = 0; j < 4; ++j)
            acc[i][j] = (v8f){0.f, 0.f, 0.f, 0.f, 0.f, 0.f, 0.f, 0.f};

    for (int kt = 0, it = 0; kt < DD; kt += KT, ++it) {
#if HAVE_ASYNC
        const int cur = it & 1;
        if (kt + KT < DD) issue_tile(kt + KT, cur ^ 1);   // prefetch next tile via ASYNC engine
#endif
        // --- stage A tile (128 x 32) as normalized bf16 hi/lo ---
        {
            const int r = tid >> 1, half = tid & 1;
            const float rn = sRn[r];
#if HAVE_ASYNC
            const float* src = &sRaw[cur][r * KT + half * 16];
#else
            const float* src = emb + ((size_t)(b * SD + bm0 + r)) * DD + kt + half * 16;
#endif
            unsigned int* dh = (unsigned int*)&sAh[r * LDSP + half * 16];
            unsigned int* dl = (unsigned int*)&sAl[r * LDSP + half * 16];
            #pragma unroll
            for (int j = 0; j < 4; ++j) {
                float4 v4 = *(const float4*)(src + 4 * j);
                float x0 = v4.x * rn, x1 = v4.y * rn, x2 = v4.z * rn, x3 = v4.w * rn;
                x0 = (x0 == x0) ? x0 : 0.f;
                x1 = (x1 == x1) ? x1 : 0.f;
                x2 = (x2 == x2) ? x2 : 0.f;
                x3 = (x3 == x3) ? x3 : 0.f;
                unsigned short h0, l0, h1, l1, h2, l2, h3, l3;
                split_bf16(x0, h0, l0); split_bf16(x1, h1, l1);
                split_bf16(x2, h2, l2); split_bf16(x3, h3, l3);
                dh[2 * j]     = (unsigned int)h0 | ((unsigned int)h1 << 16);
                dh[2 * j + 1] = (unsigned int)h2 | ((unsigned int)h3 << 16);
                dl[2 * j]     = (unsigned int)l0 | ((unsigned int)l1 << 16);
                dl[2 * j + 1] = (unsigned int)l2 | ((unsigned int)l3 << 16);
            }
#if !HAVE_ASYNC
            if (kt + KT < DD) __builtin_prefetch(src + KT, 0, 3);  // global_prefetch_b8
#endif
        }
        // --- stage B tile (32 x 128) = shifted windows of circular ahat (no circ tensor!) ---
        {
            const int n = tid >> 1, half = tid & 1;
            const int base = kt + half * 16 - bn0 - n + DD;   // in [1, 2047]
            #pragma unroll
            for (int j = 0; j < 16; ++j) {
                sBh[n * LDSP + half * 16 + j] = sCh[base + j];
                sBl[n * LDSP + half * 16 + j] = sCl[base + j];
            }
        }
        __syncthreads();

        // --- fragments in ISA VGPR layouts + WMMA ---
        FragB ah[2], al[2];
        #pragma unroll
        for (int tm = 0; tm < 2; ++tm) {
            const int m  = wm * 32 + tm * 16 + lr;
            const int kh = lh * 8;
            ah[tm].q[0] = *(const uint4*)&sAh[m * LDSP + kh];
            ah[tm].q[1] = *(const uint4*)&sAh[m * LDSP + 16 + kh];
            al[tm].q[0] = *(const uint4*)&sAl[m * LDSP + kh];
            al[tm].q[1] = *(const uint4*)&sAl[m * LDSP + 16 + kh];
        }
        #pragma unroll
        for (int tn = 0; tn < 4; ++tn) {
            const int n  = wn * 64 + tn * 16 + lr;
            const int kb = lh * 8;
            FragB bh, bl;
            bh.q[0] = *(const uint4*)&sBh[n * LDSP + kb];
            bh.q[1] = *(const uint4*)&sBh[n * LDSP + kb + 16];
            bl.q[0] = *(const uint4*)&sBl[n * LDSP + kb];
            bl.q[1] = *(const uint4*)&sBl[n * LDSP + kb + 16];
            #pragma unroll
            for (int tm = 0; tm < 2; ++tm) {
                acc[tm][tn] = __builtin_amdgcn_wmma_f32_16x16x32_bf16(
                    false, ah[tm].v, false, bh.v, (short)0, acc[tm][tn], false, false);
                acc[tm][tn] = __builtin_amdgcn_wmma_f32_16x16x32_bf16(
                    false, ah[tm].v, false, bl.v, (short)0, acc[tm][tn], false, false);
                acc[tm][tn] = __builtin_amdgcn_wmma_f32_16x16x32_bf16(
                    false, al[tm].v, false, bh.v, (short)0, acc[tm][tn], false, false);
            }
        }
#if HAVE_ASYNC
        if (kt + KT < DD) __builtin_amdgcn_s_wait_asynccnt(0);  // next raw tile landed
#endif
        __syncthreads();
    }

    // ---------- Epilogue: C/D layout (VGPR v -> M = v + 8*lane_half) ----------
    float* outp = out + (size_t)b * SD * DD;
    #pragma unroll
    for (int tm = 0; tm < 2; ++tm) {
        #pragma unroll
        for (int tn = 0; tn < 4; ++tn) {
            const int n = bn0 + wn * 64 + tn * 16 + lr;
            #pragma unroll
            for (int v = 0; v < 8; ++v) {
                const int m = bm0 + wm * 32 + tm * 16 + v + lh * 8;
                outp[(size_t)m * DD + n] = acc[tm][tn][v];
            }
        }
    }
}

extern "C" void kernel_launch(void* const* d_in, const int* in_sizes, int n_in,
                              void* d_out, int out_size, void* d_ws, size_t ws_size,
                              hipStream_t stream) {
    (void)in_sizes; (void)n_in; (void)d_ws; (void)ws_size; (void)out_size;
    const float* emb    = (const float*)d_in[0];   // (32, 512, 1024) f32
    const float* aspect = (const float*)d_in[1];   // (32, 1024) f32
    float* out = (float*)d_out;                    // (32, 512, 1024) f32

    dim3 grid(SD / BM, DD / BN, NB);               // (4, 8, 32) = 1024 blocks
    dim3 block(256);                               // 8 wave32 waves
    wordaspect_circgemm<<<grid, block, 0, stream>>>(emb, aspect, out);
}
